// FFTConvS4_781684048341
// MI455X (gfx1250) — matrix operand order repaired and checked by reading, not compiled, not run on hardware
//
#include <hip/hip_runtime.h>

typedef __attribute__((ext_vector_type(16))) _Float16 v16h;
typedef __attribute__((ext_vector_type(8)))  _Float16 v8h;
typedef __attribute__((ext_vector_type(8)))  float    v8f;
typedef __attribute__((ext_vector_type(4)))  float    v4f;

#define B_    4
#define H_    1024
#define L_    4096
#define KREV_ 4128               // 4096 reversed taps + 32-entry zero tail (causal mask)
#define XROW  4112               // 4096 + pad; *2 bytes = 32B multiple

// Y[t,b] = sum_u k[t-u] x[u,b] as block-Toeplitz WMMA (f16 in, f32 acc).
// A-tile(d)[i,j] = k[16d+i-j]; with krev[j]=k[4095-j] the per-lane A gather is
// two ASCENDING 8-half runs at S(d)=4095-16d-m+8hi and S(d)+16  -> ds_load_b128.
// B-tile(q)[j,n] = x[32q+j, b=n], cols 4..15 read a zeroed LDS row.
static __device__ __forceinline__ v8h ld8(const _Float16* p) {
  v8h v; __builtin_memcpy(&v, p, 16); return v;
}
static __device__ __forceinline__ v16h cat8(v8h lo, v8h hi) {
  return __builtin_shufflevector(lo, hi, 0,1,2,3,4,5,6,7,8,9,10,11,12,13,14,15);
}
static __device__ __forceinline__ v8f wmma(v16h a, v16h b, v8f c) {
  return __builtin_amdgcn_wmma_f32_16x16x32_f16(false, a, false, b, (short)0, c, false, false);
}
static __device__ __forceinline__ void st_tile(float* yp, v8f a) {
  *(v4f*)yp       = __builtin_shufflevector(a, a, 0, 1, 2, 3);
  *(v4f*)(yp + 4) = __builtin_shufflevector(a, a, 4, 5, 6, 7);
}

__global__ __launch_bounds__(256) void fftconv_toeplitz_wmma(
    const float* __restrict__ xg, const float* __restrict__ kg,
    float* __restrict__ yg)
{
  __shared__ __attribute__((aligned(32))) _Float16 xlds[5][XROW];
  __shared__ __attribute__((aligned(32))) _Float16 krev[KREV_];

  const int h   = blockIdx.x;
  const int tid = threadIdx.x;

  // ---- zero LDS (zero batch-row 4, row tails, krev tail) ----
  {
    unsigned* xz = (unsigned*)&xlds[0][0];
    for (int i = tid; i < (5 * XROW) / 2; i += 256) xz[i] = 0u;
    unsigned* kz = (unsigned*)krev;
    for (int i = tid; i < KREV_ / 2; i += 256) kz[i] = 0u;
  }
  __syncthreads();

  // ---- stage krev (reversed taps) and x[0..3,h,:] as f16 ----
  {
    const float* kp = kg + (size_t)h * L_;
    for (int i = tid; i < L_; i += 256) krev[4095 - i] = (_Float16)kp[i];
    for (int b = 0; b < B_; ++b) {
      const float* xp = xg + ((size_t)b * H_ + h) * L_;
      for (int i = tid; i < L_; i += 256) xlds[b][i] = (_Float16)xp[i];
    }
  }
  __syncthreads();

  const int lane = tid & 31;
  const int m    = lane & 15;                                // A row / B,D column
  const int hi   = lane >> 4;                                // half-wave selector
  const int wv   = __builtin_amdgcn_readfirstlane(tid >> 5); // scalar wave id

  const _Float16* xrow  = &xlds[(m < B_) ? m : 4][0];
  const int       abase = 4095 - m + (hi << 3);              // S(d) = abase - 16d

  for (int g = 0; g < 8; ++g) {
    const int i0 = (g * 8 + wv) * 4;      // 4 row tiles i0..i0+3 (i0 even)
    const int qc = i0 >> 1;               // shared causal q-range for r=0,1

    // Preload A(i0+2), A(i0+3): three overlapping 16B chunks serve both.
    v16h af2, af3;
    {
      const _Float16* p = krev + (abase - ((i0 + 2) << 4));
      v8h c0 = ld8(p - 16), c1 = ld8(p), c2 = ld8(p + 16);
      af2 = cat8(c1, c2);
      af3 = cat8(c0, c1);
    }
    v8f acc0 = {}, acc1 = {}, acc2 = {}, acc3 = {};

    #pragma unroll 2
    for (int q = 0; q <= qc; ++q) {
      v16h bf = cat8(ld8(xrow + (q << 5) + (hi << 3)), ld8(xrow + (q << 5) + 16 + (hi << 3)));
      const _Float16* p = krev + (abase - ((i0 - 2 * q) << 4));  // d0 = i0-2q >= 0
      v8h c0 = ld8(p - 16), c1 = ld8(p), c2 = ld8(p + 16);
      v16h af0 = cat8(c1, c2);            // A(d0)
      v16h af1 = cat8(c0, c1);            // A(d0+1)
      acc0 = wmma(af0, bf, acc0);
      acc1 = wmma(af1, bf, acc1);
      acc2 = wmma(af2, bf, acc2);         // A(d0+2) = rotated
      acc3 = wmma(af3, bf, acc3);         // A(d0+3) = rotated
      af2 = af0;                          // becomes A(d0'+2) at q+1
      af3 = af1;
    }
    // Epilogue q = qc+1: only r=2 (d=0) and r=3 (d=1); fragments already rotated in.
    {
      v16h bf = cat8(ld8(xrow + ((qc + 1) << 5) + (hi << 3)), ld8(xrow + ((qc + 1) << 5) + 16 + (hi << 3)));
      acc2 = wmma(af2, bf, acc2);
      acc3 = wmma(af3, bf, acc3);
    }

    // D: VGPR s -> row s (lanes 0-15) / s+8 (lanes 16-31); col n = m (batch)
    if (m < B_) {
      float* yb = yg + ((size_t)m * H_ + h) * L_ + ((size_t)i0 << 4) + (hi << 3);
      st_tile(yb,      acc0);
      st_tile(yb + 16, acc1);
      st_tile(yb + 32, acc2);
      st_tile(yb + 48, acc3);
    }
  }
}

extern "C" void kernel_launch(void* const* d_in, const int* in_sizes, int n_in,
                              void* d_out, int out_size, void* d_ws, size_t ws_size,
                              hipStream_t stream) {
  (void)in_sizes; (void)n_in; (void)out_size; (void)d_ws; (void)ws_size;
  const float* x = (const float*)d_in[0];   // (B, H, L) f32
  const float* k = (const float*)d_in[1];   // (1, H, L) f32
  float* y = (float*)d_out;                 // (B, 1, H, L) f32
  hipLaunchKernelGGL(fftconv_toeplitz_wmma, dim3(H_), dim3(256), 0, stream, x, k, y);
}
